// LiMNet_5617817223819
// MI455X (gfx1250) — matrix operand order repaired and checked by reading, not compiled, not run on hardware
//
#include <hip/hip_runtime.h>

#define BATCH 32
#define SEQ   512
#define NUSR  10000
#define NITM  10000
#define EDIM  128
#define E2    256
#define E3    384
#define BPB   8                 // batches per block -> A-tile M = 16 (8 xu + 8 xi)
#define NBLK  (BATCH / BPB)     // 4 blocks
#define THREADS 256             // 8 waves; wave w owns gate columns [16w,16w+16)

typedef __attribute__((ext_vector_type(16))) _Float16 v16h;
typedef __attribute__((ext_vector_type(2)))  _Float16 h2;
typedef __attribute__((ext_vector_type(8)))  float    v8f;

union AFrag { v16h v; h2 p[8]; };

struct __align__(32) Smem {
    _Float16 ueh[BPB * EDIM];   // gathered user embeddings (f16)
    _Float16 ieh[BPB * EDIM];   // gathered item embeddings (f16)
    const float* uptr[BPB];     // per-batch gather source pointers
    const float* iptr[BPB];
};

__global__ void limnet_init_last(int* p, int n) {
    int i = blockIdx.x * blockDim.x + threadIdx.x;
    if (i < n) p[i] = -1;
}

__device__ __forceinline__ float sigmoidf_fast(float x) {
    return 1.0f / (1.0f + __expf(-x));
}

// Load one B fragment (16 contiguous K-halves for this lane) from global f32,
// converting to f16.  Contiguous K order (same as A): lanes 0-15 hold K=0..15 of the k-tile,
// lanes 16-31 hold K=16..31; halves ascend in K.
__device__ __forceinline__ v16h load_bfrag_f32(const float* __restrict__ base) {
    v16h r;
    #pragma unroll
    for (int i = 0; i < 16; ++i) r[i] = (_Float16)base[i];
    return r;
}

__global__ void __launch_bounds__(THREADS, 1)
limnet_kernel(const int* __restrict__ users, const int* __restrict__ items,
              const float* __restrict__ umem0, const float* __restrict__ imem0,
              const float* __restrict__ w_ih, const float* __restrict__ w_hh,
              const float* __restrict__ b_ih, const float* __restrict__ b_hh,
              float* __restrict__ out, int* __restrict__ last_u,
              int* __restrict__ last_i)
{
    __shared__ Smem sm;

    const int tid  = threadIdx.x;
    const int b0   = blockIdx.x * BPB;
    const int lane = tid & 31;
    const int wv   = tid >> 5;      // wave id 0..7 -> column group
    const int hi   = lane >> 4;     // lane half (A K-group / B K-group / D row-group)
    const int nn   = lane & 15;     // A row (M) / B,D column (N) within tile
    const size_t iout_off = (size_t)BATCH * SEQ * EDIM;

    // ---- hoist ALL weight B-fragments into registers (constant over 512 steps)
    // Bih[kt][g]: w_ih tile at rows N = g*128 + 16*wv + nn, cols kt*32 + hi*16 + [0,16)
    v16h Bih0[3], Bih1[3], Bih2[3], Bih3[3], Bih4[3], Bih5[3], Bih6[3], Bih7[3];
    v16h Bhh0[3], Bhh1[3], Bhh2[3], Bhh3[3];
    {
        const int ncol = 16 * wv + nn;
        #pragma unroll
        for (int g = 0; g < 3; ++g) {
            const float* wr = w_ih + (size_t)(g * 128 + ncol) * E2 + hi * 16;
            Bih0[g] = load_bfrag_f32(wr + 0 * 32);
            Bih1[g] = load_bfrag_f32(wr + 1 * 32);
            Bih2[g] = load_bfrag_f32(wr + 2 * 32);
            Bih3[g] = load_bfrag_f32(wr + 3 * 32);
            Bih4[g] = load_bfrag_f32(wr + 4 * 32);
            Bih5[g] = load_bfrag_f32(wr + 5 * 32);
            Bih6[g] = load_bfrag_f32(wr + 6 * 32);
            Bih7[g] = load_bfrag_f32(wr + 7 * 32);
            const float* hr = w_hh + (size_t)(g * 128 + ncol) * EDIM + hi * 16;
            Bhh0[g] = load_bfrag_f32(hr + 0 * 32);
            Bhh1[g] = load_bfrag_f32(hr + 1 * 32);
            Bhh2[g] = load_bfrag_f32(hr + 2 * 32);
            Bhh3[g] = load_bfrag_f32(hr + 3 * 32);
        }
    }

    // per-lane bias values (constant over rows -> splat into accumulators)
    const int ncol = 16 * wv + nn;
    const float br_i = b_ih[      ncol];
    const float bz_i = b_ih[128 + ncol];
    const float bn_i = b_ih[256 + ncol];
    const float br_h = b_hh[      ncol];
    const float bz_h = b_hh[128 + ncol];
    const float bn_h = b_hh[256 + ncol];

    // A-fragment row pointers: M = nn; rows 0..7 = xu(batch nn), rows 8..15 = xi(batch nn-8)
    const _Float16* xrow_lo = (nn < BPB) ? (sm.ueh + nn * EDIM) : (sm.ieh + (nn - BPB) * EDIM); // k in [0,128)
    const _Float16* xrow_hi = (nn < BPB) ? (sm.ieh + nn * EDIM) : (sm.ueh + (nn - BPB) * EDIM); // k in [128,256)

    for (int t = 0; t < SEQ; ++t) {
        // ---- phase A: resolve gather sources via last-write table ----------
        if (tid < BPB) {
            const int bb = tid, b = b0 + bb;
            const int u  = users[b * SEQ + t];
            const int it = items[b * SEQ + t];
            const int lu = last_u[b * NUSR + u];  last_u[b * NUSR + u]  = t;
            const int li = last_i[b * NITM + it]; last_i[b * NITM + it] = t;
            sm.uptr[bb] = (lu >= 0) ? (out + (size_t)(b * SEQ + lu) * EDIM)
                                    : (umem0 + ((size_t)b * NUSR + u) * EDIM);
            sm.iptr[bb] = (li >= 0) ? (out + iout_off + (size_t)(b * SEQ + li) * EDIM)
                                    : (imem0 + ((size_t)b * NITM + it) * EDIM);
        }
        __syncthreads();

        // ---- phase B: gather embeddings -> LDS (f32 -> f16) -----------------
        for (int idx = tid; idx < BPB * EDIM; idx += THREADS) {
            const int bb = idx >> 7, c = idx & (EDIM - 1);
            sm.ueh[idx] = (_Float16)sm.uptr[bb][c];
            sm.ieh[idx] = (_Float16)sm.iptr[bb][c];
        }
        __syncthreads();

        // ---- phase C: WMMA. gi = X @ w_ih^T + b_ih ; gh = H @ w_hh^T + b_hh -
        v8f air, aiz, ain, ahr, ahz, ahn;
        #pragma unroll
        for (int j = 0; j < 8; ++j) {
            air[j] = br_i; aiz[j] = bz_i; ain[j] = bn_i;
            ahr[j] = br_h; ahz[j] = bz_h; ahn[j] = bn_h;
        }

        // input-hidden: K = 256 -> 8 k-tiles of 32 (B fragments already in regs)
        #pragma unroll
        for (int kt = 0; kt < 8; ++kt) {
            const _Float16* rp = (kt < 4) ? xrow_lo : xrow_hi;
            const int kk = (kt & 3) * 32;
            AFrag af;
            #pragma unroll
            for (int v = 0; v < 8; ++v) {
                const int kb = 2 * v + hi * 16; // contiguous K order: k = 16*hi + s (matches B)
                af.p[v] = *(const h2*)(rp + kk + kb);
            }
            const v16h* Bk = (kt == 0) ? Bih0 : (kt == 1) ? Bih1 : (kt == 2) ? Bih2 :
                             (kt == 3) ? Bih3 : (kt == 4) ? Bih4 : (kt == 5) ? Bih5 :
                             (kt == 6) ? Bih6 : Bih7;
            air = __builtin_amdgcn_wmma_f32_16x16x32_f16(false, af.v, false, Bk[0], (short)0, air, false, false);
            aiz = __builtin_amdgcn_wmma_f32_16x16x32_f16(false, af.v, false, Bk[1], (short)0, aiz, false, false);
            ain = __builtin_amdgcn_wmma_f32_16x16x32_f16(false, af.v, false, Bk[2], (short)0, ain, false, false);
        }
        // hidden-hidden: H rows = ue (m<8) / ie (m>=8) == xrow_lo; K = 128 -> 4 k-tiles
        #pragma unroll
        for (int kt = 0; kt < 4; ++kt) {
            const int kk = kt * 32;
            AFrag af;
            #pragma unroll
            for (int v = 0; v < 8; ++v) {
                const int kb = 2 * v + hi * 16;
                af.p[v] = *(const h2*)(xrow_lo + kk + kb);
            }
            const v16h* Bk = (kt == 0) ? Bhh0 : (kt == 1) ? Bhh1 : (kt == 2) ? Bhh2 : Bhh3;
            ahr = __builtin_amdgcn_wmma_f32_16x16x32_f16(false, af.v, false, Bk[0], (short)0, ahr, false, false);
            ahz = __builtin_amdgcn_wmma_f32_16x16x32_f16(false, af.v, false, Bk[1], (short)0, ahz, false, false);
            ahn = __builtin_amdgcn_wmma_f32_16x16x32_f16(false, af.v, false, Bk[2], (short)0, ahn, false, false);
        }

        // ---- phase D: GRU combine in registers, scatter = write d_out -------
        #pragma unroll
        for (int j = 0; j < 8; ++j) {
            const int m = j + hi * 8;             // D layout: VGPR j, lane>=16 -> M=j+8
            const float hval = (float)((m < BPB) ? sm.ueh[m * EDIM + ncol]
                                                 : sm.ieh[(m - BPB) * EDIM + ncol]);
            const float r  = sigmoidf_fast(air[j] + ahr[j]);
            const float z  = sigmoidf_fast(aiz[j] + ahz[j]);
            const float ng = tanhf(ain[j] + r * ahn[j]);
            const float o  = (1.0f - z) * ng + z * hval;
            const int bb = (m < BPB) ? m : (m - BPB);
            const size_t base = (m < BPB) ? (size_t)0 : iout_off;
            out[base + (size_t)((b0 + bb) * SEQ + t) * EDIM + ncol] = o;
        }
        __threadfence();    // make this step's out-rows visible to next gather
        __syncthreads();
    }
}

extern "C" void kernel_launch(void* const* d_in, const int* in_sizes, int n_in,
                              void* d_out, int out_size, void* d_ws, size_t ws_size,
                              hipStream_t stream) {
    const int*   users = (const int*)d_in[0];
    const int*   items = (const int*)d_in[1];
    const float* umem0 = (const float*)d_in[2];
    const float* imem0 = (const float*)d_in[3];
    const float* w_ih  = (const float*)d_in[4];
    const float* w_hh  = (const float*)d_in[5];
    const float* b_ih  = (const float*)d_in[6];
    const float* b_hh  = (const float*)d_in[7];
    float* out = (float*)d_out;

    int* last_u = (int*)d_ws;
    int* last_i = last_u + BATCH * NUSR;
    const int ninit = BATCH * (NUSR + NITM);

    limnet_init_last<<<(ninit + 255) / 256, 256, 0, stream>>>(last_u, ninit);
    limnet_kernel<<<NBLK, THREADS, 0, stream>>>(
        users, items, umem0, imem0, w_ih, w_hh, b_ih, b_hh, out, last_u, last_i);
}
